// gs_encoder_74818330296425
// MI455X (gfx1250) — matrix operand order repaired and checked by reading, not compiled, not run on hardware
//
#include <hip/hip_runtime.h>

#define B_    4
#define NPT   8192
#define NS    1024
#define CINC  10
#define EMB   512
#define FEATC 384

typedef __attribute__((ext_vector_type(16))) _Float16 v16h;
typedef __attribute__((ext_vector_type(8)))  float    v8f;

// ---------------------------------------------------------------------------
// 1) Farthest point sampling: one block per batch, dist[] in LDS, serial scan
// ---------------------------------------------------------------------------
__global__ __launch_bounds__(256)
void fps_kernel(const float* __restrict__ pts, const float* __restrict__ aff,
                const int* __restrict__ mask, float* __restrict__ spts,
                float* __restrict__ saff)
{
    __shared__ float dist[NPT];
    __shared__ float rv[256];
    __shared__ int   ri[256];
    __shared__ int   s_far;
    const int b = blockIdx.x, t = threadIdx.x;
    const float* P = pts + (size_t)b * NPT * CINC;
    const int*   M = mask + (size_t)b * NPT;

    int first = NPT;
    for (int i = t; i < NPT; i += 256) {
        const bool v = (M[i] > 0);
        dist[i] = v ? 1.0e10f : -3.0e38f;
        if (v && i < first) first = i;
    }
    ri[t] = first;
    __syncthreads();
    for (int k = 128; k > 0; k >>= 1) {
        if (t < k && ri[t + k] < ri[t]) ri[t] = ri[t + k];
        __syncthreads();
    }
    if (t == 0) s_far = (ri[0] < NPT) ? ri[0] : 0;
    __syncthreads();

    for (int s = 0; s < NS; ++s) {
        const int far = s_far;
        const float px = P[far * CINC + 0];
        const float py = P[far * CINC + 1];
        const float pz = P[far * CINC + 2];
        if (t < CINC) spts[((size_t)b * NS + s) * CINC + t] = P[far * CINC + t];
        if (t == CINC) saff[(size_t)b * NS + s] = aff[(size_t)b * NPT + far];

        float bv = -3.0e38f; int bi = 0;
        for (int i = t; i < NPT; i += 256) {
            const float dx = P[i * CINC + 0] - px;
            const float dy = P[i * CINC + 1] - py;
            const float dz = P[i * CINC + 2] - pz;
            const float d  = dx * dx + dy * dy + dz * dz;
            float cur = dist[i];
            if (d < cur) { cur = d; dist[i] = d; }
            if (cur > bv) { bv = cur; bi = i; }
        }
        rv[t] = bv; ri[t] = bi;
        __syncthreads();
        for (int k = 128; k > 0; k >>= 1) {
            if (t < k) {
                if (rv[t + k] > rv[t] || (rv[t + k] == rv[t] && ri[t + k] < ri[t])) {
                    rv[t] = rv[t + k]; ri[t] = ri[t + k];
                }
            }
            __syncthreads();
        }
        if (t == 0) s_far = ri[0];
        __syncthreads();
    }
}

// ---------------------------------------------------------------------------
// 2) Threshold selection: th=0.5 unless no element > 0.5, then k-th largest
// ---------------------------------------------------------------------------
__global__ __launch_bounds__(256)
void thresh_kernel(const float* __restrict__ a, int N, int kmin,
                   float* __restrict__ th_out)
{
    __shared__ int cnt;
    const int b = blockIdx.x, t = threadIdx.x;
    const float* A = a + (size_t)b * N;
    if (t == 0) cnt = 0;
    __syncthreads();
    int c = 0;
    for (int i = t; i < N; i += 256) c += (A[i] > 0.5f) ? 1 : 0;
    atomicAdd(&cnt, c);
    __syncthreads();
    if (cnt > 0) { if (t == 0) th_out[b] = 0.5f; return; }
    float lo = 0.0f, hi = 1.0f;
    for (int it = 0; it < 32; ++it) {
        const float mid = 0.5f * (lo + hi);
        __syncthreads();
        if (t == 0) cnt = 0;
        __syncthreads();
        int lc = 0;
        for (int i = t; i < N; i += 256) lc += (A[i] >= mid) ? 1 : 0;
        atomicAdd(&cnt, lc);
        __syncthreads();
        if (cnt >= kmin) lo = mid; else hi = mid;
    }
    if (t == 0) th_out[b] = lo;
}

// bin mask (+ optional masked 10-channel product)
__global__ __launch_bounds__(256)
void binmask_kernel(const float* __restrict__ a, const float* __restrict__ th,
                    const float* __restrict__ src10, float* __restrict__ bin,
                    float* __restrict__ masked10, int N)
{
    const size_t i = (size_t)blockIdx.x * 256 + threadIdx.x;
    if (i >= (size_t)B_ * N) return;
    const int b = (int)(i / N);
    const float m = (a[i] >= th[b]) ? 1.0f : 0.0f;
    bin[i] = m;
    if (masked10) {
        for (int c = 0; c < CINC; ++c)
            masked10[i * CINC + c] = src10[i * CINC + c] * m;
    }
}

__global__ __launch_bounds__(256)
void mask2f_kernel(const int* __restrict__ m, float* __restrict__ mf, int total)
{
    const int i = blockIdx.x * 256 + threadIdx.x;
    if (i < total) mf[i] = (m[i] > 0) ? 1.0f : 0.0f;
}

// ---------------------------------------------------------------------------
// 3) kNN grouping: one block per (query, batch); dist[] in LDS; k argmin passes
//    out layout (B, 10, S, K).  qsc = bin_s value scales both query and output.
// ---------------------------------------------------------------------------
__global__ __launch_bounds__(256)
void knn_kernel(const float* __restrict__ spts, const float* __restrict__ bin_s,
                const float* __restrict__ src, const float* __restrict__ vmask,
                float* __restrict__ out, int K)
{
    __shared__ float dist[NPT];
    __shared__ float rv[256];
    __shared__ int   ri[256];
    const int s = blockIdx.x, b = blockIdx.y, t = threadIdx.x;
    const float* Q = spts + ((size_t)b * NS + s) * CINC;
    const float qsc = bin_s ? bin_s[(size_t)b * NS + s] : 1.0f;
    const float qx = Q[0] * qsc, qy = Q[1] * qsc, qz = Q[2] * qsc;
    const float* P = src + (size_t)b * NPT * CINC;
    const float* V = vmask + (size_t)b * NPT;

    for (int i = t; i < NPT; i += 256) {
        float d = 3.0e30f;
        if (V[i] > 0.0f) {
            const float dx = P[i * CINC + 0] - qx;
            const float dy = P[i * CINC + 1] - qy;
            const float dz = P[i * CINC + 2] - qz;
            d = dx * dx + dy * dy + dz * dz;
        }
        dist[i] = d;
    }
    __syncthreads();
    for (int j = 0; j < K; ++j) {
        float bv = 3.0e38f; int bi = 0;
        for (int i = t; i < NPT; i += 256) {
            const float v = dist[i];
            if (v < bv) { bv = v; bi = i; }
        }
        rv[t] = bv; ri[t] = bi;
        __syncthreads();
        for (int k = 128; k > 0; k >>= 1) {
            if (t < k) {
                if (rv[t + k] < rv[t] || (rv[t + k] == rv[t] && ri[t + k] < ri[t])) {
                    rv[t] = rv[t + k]; ri[t] = ri[t + k];
                }
            }
            __syncthreads();
        }
        const int sel = ri[0];
        if (t < CINC)
            out[(((size_t)b * CINC + t) * NS + s) * K + j] = P[(size_t)sel * CINC + t] * qsc;
        __syncthreads();
        if (t == 0) dist[sel] = 3.0e38f;
        __syncthreads();
    }
}

// ---------------------------------------------------------------------------
// 4) conv1x1 as WMMA GEMM.  4 waves / block; each wave owns a 16(M) x 64(N)
//    strip: the A (weight) fragment is loaded once per K-step and reused by
//    4 WMMAs.  f32 -> f16 fragments per ISA 7.12.2 layouts, f32 accumulate.
// ---------------------------------------------------------------------------
__global__ __launch_bounds__(128)
void conv_wmma_kernel(const float* __restrict__ Wt, const float* __restrict__ bias,
                      const float* __restrict__ act, const float* __restrict__ colmask,
                      float* __restrict__ out, int Cin, int Cout, int SK, int Kk)
{
    const int tid  = threadIdx.x;
    const int wv   = tid >> 5;          // wave 0..3 within block
    const int lane = tid & 31;
    const int hf   = lane >> 4;         // lane half
    const int idx  = lane & 15;
    const int b     = blockIdx.z;
    const int mbase = blockIdx.y << 4;
    const int nwave = (blockIdx.x << 8) + (wv << 6);   // 64 columns per wave
    const float* A  = act + (size_t)b * Cin * SK;
    const float* Wr = Wt + (size_t)(mbase + idx) * Cin;

    int   cols[4];
    float cmv[4];
#pragma unroll
    for (int sub = 0; sub < 4; ++sub) {
        cols[sub] = nwave + (sub << 4) + idx;
        cmv[sub]  = colmask ? colmask[(size_t)b * NS + cols[sub] / Kk] : 1.0f;
    }

    v8f cc[4] = {};
    const int Ksteps = (Cin + 31) >> 5;
    for (int ks = 0; ks < Ksteps; ++ks) {
        const int kb = ks << 5;
        v16h av = {};
#pragma unroll
        for (int v = 0; v < 8; ++v) {
            // A fragment (weights): K = 16*(v>>2) + 8*half + 2*(v&3)
            const int ka = kb + ((v >> 2) << 4) + (hf << 3) + ((v & 3) << 1);
            av[2 * v]     = (_Float16)((ka     < Cin) ? Wr[ka]     : 0.0f);
            av[2 * v + 1] = (_Float16)((ka + 1 < Cin) ? Wr[ka + 1] : 0.0f);
        }
#pragma unroll
        for (int sub = 0; sub < 4; ++sub) {
            v16h bv = {};
#pragma unroll
            for (int v = 0; v < 8; ++v) {
                // B fragment (activations): K = 16*(v>>2) + 8*half + 2*(v&3), column per lane
                const int kr = kb + ((v >> 2) << 4) + (hf << 3) + ((v & 3) << 1);
                bv[2 * v] = (_Float16)((kr < Cin)
                    ? A[(size_t)kr * SK + cols[sub]] * cmv[sub] : 0.0f);
                bv[2 * v + 1] = (_Float16)((kr + 1 < Cin)
                    ? A[(size_t)(kr + 1) * SK + cols[sub]] * cmv[sub] : 0.0f);
            }
            cc[sub] = __builtin_amdgcn_wmma_f32_16x16x32_f16(
                false, av, false, bv, (short)0, cc[sub], false, false);
        }
        // uniform-guarded prefetch of next K-step activation lines
        if (kb + 32 < Cin)
            __builtin_prefetch(&A[(size_t)(kb + 32 + (hf << 4)) * SK + cols[0]], 0, 1);
    }
#pragma unroll
    for (int sub = 0; sub < 4; ++sub) {
#pragma unroll
        for (int r = 0; r < 8; ++r) {
            const int m = mbase + r + (hf << 3);
            out[((size_t)b * Cout + m) * SK + cols[sub]] = cc[sub][r] + bias[m];
        }
    }
}

// BN statistics (optionally mask-weighted): mean + invstd per channel
__global__ __launch_bounds__(256)
void bn_stats_kernel(const float* __restrict__ x, const float* __restrict__ am,
                     float* __restrict__ mean, float* __restrict__ invstd,
                     int Cout, int SK, int Kk)
{
    __shared__ float s0[256], s1[256], s2[256];
    const int cch = blockIdx.x, t = threadIdx.x;
    float sw = 0.0f, sx = 0.0f, sxx = 0.0f;
    for (int b = 0; b < B_; ++b) {
        const float* X = x + ((size_t)b * Cout + cch) * SK;
        for (int i = t; i < SK; i += 256) {
            const float w = am ? am[(size_t)b * NS + i / Kk] : 1.0f;
            const float v = X[i];
            sw += w; sx += v * w; sxx += v * v * w;
        }
    }
    s0[t] = sw; s1[t] = sx; s2[t] = sxx;
    __syncthreads();
    for (int k = 128; k > 0; k >>= 1) {
        if (t < k) { s0[t] += s0[t + k]; s1[t] += s1[t + k]; s2[t] += s2[t + k]; }
        __syncthreads();
    }
    if (t == 0) {
        const float cnt = s0[0] + 1e-5f;
        const float mu  = s1[0] / cnt;
        const float var = s2[0] / cnt - mu * mu;
        mean[cch]   = mu;
        invstd[cch] = rsqrtf(var + 1e-5f);
    }
}

// BN apply + ReLU into destination channels [0, Cout) of a (B, Cnext, SK) buffer
__global__ __launch_bounds__(256)
void bn_apply_kernel(const float* __restrict__ x, const float* __restrict__ mean,
                     const float* __restrict__ invstd, const float* __restrict__ bnw,
                     const float* __restrict__ bnb, float* __restrict__ dst,
                     int Cout, int Cnext, int SK)
{
    const size_t i = (size_t)blockIdx.x * 256 + threadIdx.x;
    const size_t total = (size_t)B_ * Cout * SK;
    if (i >= total) return;
    const int    col = (int)(i % SK);
    const size_t r   = i / SK;
    const int    cch = (int)(r % Cout);
    const int    b   = (int)(r / Cout);
    float v = (x[i] - mean[cch]) * invstd[cch];
    if (bnw) v = v * bnw[cch] + bnb[cch];
    v = v > 0.0f ? v : 0.0f;
    dst[((size_t)b * Cnext + cch) * SK + col] = v;
}

// copy the raw grouped 10 channels into dst channels [Cout, Cout+10)
__global__ __launch_bounds__(256)
void concat_copy_kernel(const float* __restrict__ g, float* __restrict__ dst,
                        int Cout, int Cnext, int SK)
{
    const size_t i = (size_t)blockIdx.x * 256 + threadIdx.x;
    const size_t total = (size_t)B_ * CINC * SK;
    if (i >= total) return;
    const int    col = (int)(i % SK);
    const size_t r   = i / SK;
    const int    cch = (int)(r % CINC);
    const int    b   = (int)(r / CINC);
    dst[((size_t)b * Cnext + Cout + cch) * SK + col] = g[i];
}

// fused BN + ReLU + max over k for the last layer -> feat (B, 384, S) at coff
__global__ __launch_bounds__(256)
void bn_max_kernel(const float* __restrict__ x, const float* __restrict__ mean,
                   const float* __restrict__ invstd, const float* __restrict__ bnw,
                   const float* __restrict__ bnb, float* __restrict__ feat,
                   int Cout, int Kk, int coff)
{
    const size_t i = (size_t)blockIdx.x * 256 + threadIdx.x;
    const size_t total = (size_t)B_ * Cout * NS;
    if (i >= total) return;
    const int    s   = (int)(i % NS);
    const size_t r   = i / NS;
    const int    cch = (int)(r % Cout);
    const int    b   = (int)(r / Cout);
    const float* X = x + (((size_t)b * Cout + cch) * NS + s) * Kk;
    const float mu = mean[cch], is = invstd[cch];
    const float w  = bnw ? bnw[cch] : 1.0f;
    const float bb = bnb ? bnb[cch] : 0.0f;
    float best = -3.0e38f;
    for (int kk = 0; kk < Kk; ++kk) {
        float v = (X[kk] - mu) * is * w + bb;
        v = v > 0.0f ? v : 0.0f;
        if (v > best) best = v;
    }
    feat[((size_t)b * FEATC + coff + cch) * NS + s] = best;
}

// ---------------------------------------------------------------------------
// 5) Embedding GEMM via WMMA: (B*S, 384) x (384, 512) + bias, optional row
//    mask.  4 waves / block; each wave owns 64 rows x 16 cols and reuses the
//    Wemb (B) fragment across 4 WMMAs per K-step.
// ---------------------------------------------------------------------------
__global__ __launch_bounds__(128)
void embed_wmma_kernel(const float* __restrict__ feat, const float* __restrict__ Wemb,
                       const float* __restrict__ bemb, const float* __restrict__ rowmask,
                       float* __restrict__ out)
{
    const int tid  = threadIdx.x;
    const int wv   = tid >> 5;
    const int lane = tid & 31;
    const int hf   = lane >> 4;
    const int idx  = lane & 15;
    const int nbase = blockIdx.x << 4;                 // embed columns
    const int rwave = (blockIdx.y << 8) + (wv << 6);   // 64 rows per wave

    v8f cc[4] = {};
    for (int ks = 0; ks < FEATC / 32; ++ks) {
        const int kb = ks << 5;
        v16h bv = {};
#pragma unroll
        for (int v = 0; v < 8; ++v) {
            const int kr = kb + ((v >> 2) << 4) + (hf << 3) + ((v & 3) << 1);
            bv[2 * v]     = (_Float16)Wemb[(size_t)kr * EMB + nbase + idx];
            bv[2 * v + 1] = (_Float16)Wemb[(size_t)(kr + 1) * EMB + nbase + idx];
        }
#pragma unroll
        for (int sub = 0; sub < 4; ++sub) {
            const int rowA = rwave + (sub << 4) + idx;
            const int bA = rowA / NS, sA = rowA % NS;
            v16h av = {};
#pragma unroll
            for (int v = 0; v < 8; ++v) {
                const int ka = kb + ((v >> 2) << 4) + (hf << 3) + ((v & 3) << 1);
                av[2 * v]     = (_Float16)feat[((size_t)bA * FEATC + ka) * NS + sA];
                av[2 * v + 1] = (_Float16)feat[((size_t)bA * FEATC + ka + 1) * NS + sA];
            }
            cc[sub] = __builtin_amdgcn_wmma_f32_16x16x32_f16(
                false, av, false, bv, (short)0, cc[sub], false, false);
        }
        if (kb + 32 < FEATC)
            __builtin_prefetch(&Wemb[(size_t)(kb + 32 + (hf << 4)) * EMB + nbase + idx], 0, 1);
    }
#pragma unroll
    for (int sub = 0; sub < 4; ++sub) {
#pragma unroll
        for (int r = 0; r < 8; ++r) {
            const int row = rwave + (sub << 4) + r + (hf << 3);
            float v = cc[sub][r] + bemb[nbase + idx];
            if (rowmask) v *= rowmask[row];
            out[(size_t)row * EMB + nbase + idx] = v;
        }
    }
}

__global__ __launch_bounds__(256)
void copy_kernel(const float* __restrict__ src, float* __restrict__ dst, int total)
{
    const int i = blockIdx.x * 256 + threadIdx.x;
    if (i < total) dst[i] = src[i];
}

// ---------------------------------------------------------------------------
// Host orchestration
// ---------------------------------------------------------------------------
static inline int cdiv(long long a, long long b) { return (int)((a + b - 1) / b); }

extern "C" void kernel_launch(void* const* d_in, const int* in_sizes, int n_in,
                              void* d_out, int out_size, void* d_ws, size_t ws_size,
                              hipStream_t stream)
{
    (void)in_sizes; (void)n_in; (void)out_size; (void)ws_size;
    const float* points = (const float*)d_in[0];
    const float* aff    = (const float*)d_in[1];
    const int*   maskI  = (const int*)d_in[2];
    const float* Wemb   = (const float*)d_in[3];
    const float* bemb   = (const float*)d_in[4];

    struct Layer { const float *W, *b, *bnb, *bnw; int Cout; };
    Layer L[2][3];
    const int outs[2][3] = {{32, 64, 128}, {64, 128, 256}};
    int p = 5;
    for (int i = 0; i < 2; ++i)
        for (int j = 0; j < 3; ++j) {
            L[i][j].W   = (const float*)d_in[p++];
            L[i][j].b   = (const float*)d_in[p++];
            L[i][j].bnb = (const float*)d_in[p++];
            L[i][j].bnw = (const float*)d_in[p++];
            L[i][j].Cout = outs[i][j];
        }

    // workspace layout
    float* w = (float*)d_ws;
    float* spts  = w; w += (size_t)B_ * NS * CINC;
    float* saff  = w; w += (size_t)B_ * NS;
    float* th_s  = w; w += 4;
    float* th_p  = w; w += 4;
    float* bin_s = w; w += (size_t)B_ * NS;
    float* bin_p = w; w += (size_t)B_ * NPT;
    float* maskf = w; w += (size_t)B_ * NPT;
    float* paff  = w; w += (size_t)B_ * NPT * CINC;
    float* meanb = w; w += 256;
    float* isb   = w; w += 256;
    float* pfF   = w; w += (size_t)B_ * FEATC * NS;
    float* afF   = w; w += (size_t)B_ * FEATC * NS;
    float* gp    = w; w += (size_t)B_ * CINC * NS * 32;
    float* ga    = w; w += (size_t)B_ * CINC * NS * 32;
    float* bufA  = w; w += (size_t)B_ * 160 * NS * 32;
    float* bufB  = w; w += (size_t)B_ * 160 * NS * 32;
    float* bufX  = w; w += (size_t)B_ * 256 * NS * 32;

    // ---- FPS + masks ----
    fps_kernel<<<B_, 256, 0, stream>>>(points, aff, maskI, spts, saff);
    thresh_kernel<<<B_, 256, 0, stream>>>(saff, NS, 102, th_s);   // k_min = 0.1*1024
    thresh_kernel<<<B_, 256, 0, stream>>>(aff, NPT, 819, th_p);   // k_min = 0.1*8192
    mask2f_kernel<<<cdiv((long long)B_ * NPT, 256), 256, 0, stream>>>(maskI, maskf, B_ * NPT);
    binmask_kernel<<<cdiv((long long)B_ * NS, 256), 256, 0, stream>>>(
        saff, th_s, nullptr, bin_s, nullptr, NS);
    binmask_kernel<<<cdiv((long long)B_ * NPT, 256), 256, 0, stream>>>(
        aff, th_p, points, bin_p, paff, NPT);

    // ---- per-block pipeline ----
    const int Kks[2]  = {16, 32};
    const int coff[2] = {0, 128};
    for (int i = 0; i < 2; ++i) {
        const int Kk = Kks[i];
        const int SK = NS * Kk;
        dim3 kgrid(NS, B_);
        knn_kernel<<<kgrid, 256, 0, stream>>>(spts, nullptr, points, maskf, gp, Kk);
        knn_kernel<<<kgrid, 256, 0, stream>>>(spts, bin_s, paff, bin_p, ga, Kk);

        for (int path = 0; path < 2; ++path) {
            const bool masked = (path == 1);
            const float* grouped = masked ? ga : gp;
            const float* cmask   = masked ? bin_s : nullptr;
            const float* amask   = masked ? bin_s : nullptr;
            const float* in = grouped;
            int Cin = CINC;
            for (int j = 0; j < 3; ++j) {
                const int Cout = L[i][j].Cout;
                dim3 g(SK / 256, Cout / 16, B_);   // 4 waves x 64 cols per block
                conv_wmma_kernel<<<g, 128, 0, stream>>>(
                    L[i][j].W, L[i][j].b, in, cmask, bufX, Cin, Cout, SK, Kk);
                bn_stats_kernel<<<Cout, 256, 0, stream>>>(bufX, amask, meanb, isb, Cout, SK, Kk);
                const float* bw = masked ? L[i][j].bnw : nullptr;
                const float* bb = masked ? L[i][j].bnb : nullptr;
                if (j < 2) {
                    const int Cnext = Cout + CINC;
                    float* dst = (j == 0) ? bufA : bufB;
                    bn_apply_kernel<<<cdiv((long long)B_ * Cout * SK, 256), 256, 0, stream>>>(
                        bufX, meanb, isb, bw, bb, dst, Cout, Cnext, SK);
                    concat_copy_kernel<<<cdiv((long long)B_ * CINC * SK, 256), 256, 0, stream>>>(
                        grouped, dst, Cout, Cnext, SK);
                    in = dst;
                    Cin = Cnext;
                } else {
                    float* feat = masked ? afF : pfF;
                    bn_max_kernel<<<cdiv((long long)B_ * Cout * NS, 256), 256, 0, stream>>>(
                        bufX, meanb, isb, bw, bb, feat, Cout, Kk, coff[i]);
                }
            }
        }
    }

    // ---- embedding GEMMs + outputs (pf, af, saff) ----
    float* out_pf   = (float*)d_out;
    float* out_af   = out_pf + (size_t)B_ * NS * EMB;
    float* out_saff = out_af + (size_t)B_ * NS * EMB;
    dim3 eg(EMB / 16, (B_ * NS) / 256);   // 4 waves x 64 rows per block
    embed_wmma_kernel<<<eg, 128, 0, stream>>>(pfF, Wemb, bemb, nullptr, out_pf);
    embed_wmma_kernel<<<eg, 128, 0, stream>>>(afF, Wemb, bemb, bin_s, out_af);
    copy_kernel<<<cdiv((long long)B_ * NS, 256), 256, 0, stream>>>(saff, out_saff, B_ * NS);
}
